// DotAttnConv_8074538517006
// MI455X (gfx1250) — matrix operand order repaired and checked by reading, not compiled, not run on hardware
//
#include <hip/hip_runtime.h>
#include <hip/hip_bf16.h>

// ---------------------------------------------------------------------------
// DotAttnConv for MI455X (gfx1250, wave32, WMMA + async-LDS staging)
//   N=50000 nodes, D=128, H=4 heads, HD=32, E=800000 edges
// - Fused QKV projection: one pass over xq/x, 12 v_wmma per wave.
// - Weights staged to LDS via global_load_async_to_lds_b128 (ASYNCcnt).
//   LDS offset is derived from the shared array's flat address (low 32 bits,
//   per the LDS aperture mapping) -- the ptrtoint also CAPTURES the shared
//   object so the asm "memory" clobber really orders against the ds_loads.
// - Edge softmax/scatter: float4 gathers (L2-resident) + fp32 global atomics.
// - LayerNorm: one wave32 per node, shuffle reductions.
// - Final WMMA GEMM fuses residual add + isolated-node select.
// ---------------------------------------------------------------------------

typedef __attribute__((ext_vector_type(16))) __bf16 v16bf;
typedef __attribute__((ext_vector_type(8)))  __bf16 v8bf;
typedef __attribute__((ext_vector_type(4)))  __bf16 v4bf;
typedef __attribute__((ext_vector_type(8)))  float  v8f;

#define DD   128
#define HH   4
#define HDM  32
#define LN_EPS 1e-5f
#define WBYTES (DD * DD * 2)   // one bf16 128x128 weight = 32 KB

// ----- order-preserving float <-> u32 mapping (for atomicMax on floats) -----
__device__ __forceinline__ unsigned f2ord(float f) {
    unsigned u = __float_as_uint(f);
    return (u & 0x80000000u) ? ~u : (u | 0x80000000u);
}
__device__ __forceinline__ float ord2f(unsigned u) {
    return (u & 0x80000000u) ? __uint_as_float(u ^ 0x80000000u)
                             : __uint_as_float(~u);
}
#define NEG_INF_ORD 0x007FFFFFu   // f2ord(-inf)

// ----- LDS offset of a __shared__ object from its flat address -------------
// Generic LDS addresses are {SHARED_BASE tag, offset[31:0]} (ISA 10.2), so the
// low 32 bits are the in-group LDS byte offset. The ptrtoint captures the
// object, making the async-asm "memory" clobber alias it (required!).
__device__ __forceinline__ unsigned lds_off_of(const void* p) {
    return (unsigned)(size_t)p;
}

// ----- async copy of one 16-byte chunk into LDS (raw LDS byte offset) ------
__device__ __forceinline__ void async_b128_to_lds(unsigned lds_off,
                                                  const void* gaddr) {
    asm volatile("global_load_async_to_lds_b128 %0, %1, off"
                 :: "v"(lds_off), "v"((unsigned long long)gaddr)
                 : "memory");
}
__device__ __forceinline__ void wait_async_all() {
    asm volatile("s_wait_asynccnt 0x0" ::: "memory");
}

// ----- generic fill --------------------------------------------------------
__global__ void fill_u32_k(unsigned* __restrict__ p, unsigned v, size_t n) {
    size_t i = (size_t)blockIdx.x * blockDim.x + threadIdx.x;
    if (i < n) p[i] = v;
}

// ----- xq = x + pos_emb[visit_pos]; bf16 copies of xq and x ----------------
__global__ void prep_x_k(const float* __restrict__ x,
                         const long long* __restrict__ vp,
                         const float* __restrict__ pe,
                         __bf16* __restrict__ xq_bf,
                         __bf16* __restrict__ x_bf, int N) {
    size_t i = (size_t)blockIdx.x * blockDim.x + threadIdx.x;
    if (i >= (size_t)N * DD) return;
    int n = (int)(i >> 7);
    int d = (int)(i & (DD - 1));
    float xv = x[i];
    float p  = pe[(size_t)vp[n] * DD + d];
    xq_bf[i] = (__bf16)(xv + p);
    x_bf[i]  = (__bf16)xv;
}

// ----- transpose + convert the three 128x128 weights to bf16 [out][in] -----
__global__ void prep_w_k(const float* __restrict__ Wq,
                         const float* __restrict__ Wk,
                         const float* __restrict__ Wo,
                         __bf16* __restrict__ Wqt,
                         __bf16* __restrict__ Wkt,
                         __bf16* __restrict__ Wot) {
    int i = blockIdx.x * blockDim.x + threadIdx.x;
    if (i >= DD * DD) return;
    int k = i >> 7;          // input dim (row of W)
    int n = i & (DD - 1);    // output dim (col of W)
    int t = n * DD + k;      // transposed index
    Wqt[t] = (__bf16)Wq[i];
    Wkt[t] = (__bf16)Wk[i];
    Wot[t] = (__bf16)Wo[i];
}

// ----- build 16x32 bf16 A fragment for one lane ----------------------------
__device__ __forceinline__ v16bf load_a_frag(const __bf16* A, size_t abase,
                                             int k0, int s) {
    v8bf lo = *(const v8bf*)(A + abase + k0 + s * 16);
    v8bf hi = *(const v8bf*)(A + abase + k0 + s * 16 + 8);
    v16bf a;
#pragma unroll
    for (int i = 0; i < 8; ++i) { a[i] = lo[i]; a[i + 8] = hi[i]; }
    return a;
}

// ----- fused QKV projection:  hq = xq*Wq, hk = xq*Wk, hv = x*Wk ------------
// Block = 256 thr = 8 waves; wave w owns output cols [16w,16w+16);
// block b owns rows [16b,16b+16). Wq_t/Wk_t staged to LDS asynchronously.
__global__ __launch_bounds__(256)
void gemm_qkv_k(const __bf16* __restrict__ xq, const __bf16* __restrict__ xb,
                const __bf16* __restrict__ Wqt, const __bf16* __restrict__ Wkt,
                float* __restrict__ hq, float* __restrict__ hk,
                float* __restrict__ hv, int M) {
    __shared__ __bf16 sW[2 * DD * DD];   // 64 KB: [0,32K)=Wq_t  [32K,64K)=Wk_t
    const unsigned sbase = lds_off_of(sW);

    for (unsigned i = threadIdx.x; i < WBYTES / 16; i += 256) {
        unsigned o = i * 16;
        async_b128_to_lds(sbase + o,          (const char*)Wqt + o);
        async_b128_to_lds(sbase + WBYTES + o, (const char*)Wkt + o);
    }
    wait_async_all();
    __syncthreads();

    const int lane = threadIdx.x & 31;
    const int wv   = threadIdx.x >> 5;
    const int r    = lane & 15;
    const int s    = lane >> 4;

    int arow = blockIdx.x * 16 + r;
    if (arow > M - 1) arow = M - 1;
    const size_t   abase = (size_t)arow * DD;
    const unsigned bbyte = ((unsigned)(wv * 16 + r) * DD + s * 16) * 2;
    const char*    sWb   = (const char*)sW;

    v8f accq = {}, acck = {}, accv = {};
#pragma unroll
    for (int k0 = 0; k0 < DD; k0 += 32) {
        v16bf aq = load_a_frag(xq, abase, k0, s);
        v16bf ax = load_a_frag(xb, abase, k0, s);
        v16bf bq = *(const v16bf*)(sWb + bbyte + k0 * 2);
        v16bf bk = *(const v16bf*)(sWb + WBYTES + bbyte + k0 * 2);
        accq = __builtin_amdgcn_wmma_f32_16x16x32_bf16(false, aq, false, bq,
                                                       (short)0, accq, false, false);
        acck = __builtin_amdgcn_wmma_f32_16x16x32_bf16(false, aq, false, bk,
                                                       (short)0, acck, false, false);
        accv = __builtin_amdgcn_wmma_f32_16x16x32_bf16(false, ax, false, bk,
                                                       (short)0, accv, false, false);
    }

    const int ncol = wv * 16 + r;
#pragma unroll
    for (int j = 0; j < 8; ++j) {
        const int mg = blockIdx.x * 16 + s * 8 + j;   // C/D layout: M = s*8 + j
        if (mg < M) {
            const size_t idx = (size_t)mg * DD + ncol;
            hq[idx] = accq[j];
            hk[idx] = acck[j];
            hv[idx] = accv[j];
        }
    }
}

// ----- final GEMM: out = deg>0 ? normed*Wout + x : x -----------------------
__global__ __launch_bounds__(256)
void gemm_out_k(const __bf16* __restrict__ A, const __bf16* __restrict__ Wot,
                const float* __restrict__ xres, const int* __restrict__ deg,
                float* __restrict__ outp, int M) {
    __shared__ __bf16 sW[DD * DD];       // 32 KB
    const unsigned sbase = lds_off_of(sW);

    for (unsigned i = threadIdx.x; i < WBYTES / 16; i += 256) {
        unsigned o = i * 16;
        async_b128_to_lds(sbase + o, (const char*)Wot + o);
    }
    wait_async_all();
    __syncthreads();

    const int lane = threadIdx.x & 31;
    const int wv   = threadIdx.x >> 5;
    const int r    = lane & 15;
    const int s    = lane >> 4;

    int arow = blockIdx.x * 16 + r;
    if (arow > M - 1) arow = M - 1;
    const size_t   abase = (size_t)arow * DD;
    const unsigned bbyte = ((unsigned)(wv * 16 + r) * DD + s * 16) * 2;
    const char*    sWb   = (const char*)sW;

    v8f acc = {};
#pragma unroll
    for (int k0 = 0; k0 < DD; k0 += 32) {
        v16bf a = load_a_frag(A, abase, k0, s);
        v16bf b = *(const v16bf*)(sWb + bbyte + k0 * 2);
        acc = __builtin_amdgcn_wmma_f32_16x16x32_bf16(false, a, false, b,
                                                      (short)0, acc, false, false);
    }

    const int ncol = wv * 16 + r;
#pragma unroll
    for (int j = 0; j < 8; ++j) {
        const int mg = blockIdx.x * 16 + s * 8 + j;
        if (mg < M) {
            const size_t idx = (size_t)mg * DD + ncol;
            const float  xv  = xres[idx];
            outp[idx] = (deg[mg] > 0) ? (acc[j] + xv) : xv;
        }
    }
}

// ----- per-(edge,head) scores + segment max + in-degree --------------------
__global__ __launch_bounds__(256)
void edge_scores_k(const long long* __restrict__ ei,
                   const float* __restrict__ hq, const float* __restrict__ hk,
                   float* __restrict__ scores, unsigned* __restrict__ segmax,
                   int* __restrict__ deg, int E) {
    int idx = blockIdx.x * blockDim.x + threadIdx.x;
    if (idx >= E * HH) return;
    int e = idx >> 2, h = idx & 3;
    int src = (int)ei[e];
    int dst = (int)ei[(size_t)E + e];
    const float4* q = (const float4*)(hq + (size_t)src * DD + h * HDM);
    const float4* k = (const float4*)(hk + (size_t)dst * DD + h * HDM);
    float sum = 0.f;
#pragma unroll
    for (int i = 0; i < HDM / 4; ++i) {
        float4 a = q[i], b = k[i];
        sum += a.x * b.x + a.y * b.y + a.z * b.z + a.w * b.w;
    }
    // TEMP == 1.0
    scores[idx] = sum;
    atomicMax(segmax + (size_t)dst * HH + h, f2ord(sum));
    if (h == 0) atomicAdd(deg + dst, 1);
}

// ----- ex = exp(score - segmax[dst]);  denom[dst] += ex --------------------
__global__ __launch_bounds__(256)
void edge_exp_k(const long long* __restrict__ ei,
                float* __restrict__ scores, const unsigned* __restrict__ segmax,
                float* __restrict__ denom, int E) {
    int idx = blockIdx.x * blockDim.x + threadIdx.x;
    if (idx >= E * HH) return;
    int e = idx >> 2, h = idx & 3;
    int dst = (int)ei[(size_t)E + e];
    float m  = ord2f(segmax[(size_t)dst * HH + h]);
    float ex = __expf(scores[idx] - m);
    scores[idx] = ex;                       // reuse buffer for ex
    atomicAdd(denom + (size_t)dst * HH + h, ex);
}

// ----- agg[dst] += hv[src] * alpha  (alpha = ex / denom[dst]) --------------
__global__ __launch_bounds__(256)
void edge_agg_k(const long long* __restrict__ ei,
                const float* __restrict__ scores, const float* __restrict__ denom,
                const float* __restrict__ hv, float* __restrict__ agg, int E) {
    int e = blockIdx.x * blockDim.x + threadIdx.x;
    if (e >= E) return;
    int src = (int)ei[e];
    int dst = (int)ei[(size_t)E + e];
    float alpha[HH];
#pragma unroll
    for (int h = 0; h < HH; ++h)
        alpha[h] = scores[(size_t)e * HH + h] / denom[(size_t)dst * HH + h];
    const float4* v4 = (const float4*)(hv + (size_t)src * DD);
    float* ab = agg + (size_t)dst * DD;
#pragma unroll 4
    for (int i = 0; i < DD / 4; ++i) {
        float4 v = v4[i];
        float a  = alpha[i >> 3];
        atomicAdd(ab + i * 4 + 0, v.x * a);
        atomicAdd(ab + i * 4 + 1, v.y * a);
        atomicAdd(ab + i * 4 + 2, v.z * a);
        atomicAdd(ab + i * 4 + 3, v.w * a);
    }
}

// ----- LayerNorm over 128 dims: one wave32 per node, lane holds float4 -----
__global__ __launch_bounds__(256)
void layernorm_k(const float* __restrict__ agg, const float* __restrict__ g,
                 const float* __restrict__ b, __bf16* __restrict__ normed,
                 int N) {
    const int lane = threadIdx.x & 31;
    const int node = blockIdx.x * 8 + (threadIdx.x >> 5);
    if (node >= N) return;   // wave-uniform exit
    const float4 v = *(const float4*)(agg + (size_t)node * DD + lane * 4);
    float sum = v.x + v.y + v.z + v.w;
#pragma unroll
    for (int o = 16; o > 0; o >>= 1) sum += __shfl_xor(sum, o, 32);
    const float mu = sum * (1.0f / DD);
    float4 c = { v.x - mu, v.y - mu, v.z - mu, v.w - mu };
    float sq = c.x * c.x + c.y * c.y + c.z * c.z + c.w * c.w;
#pragma unroll
    for (int o = 16; o > 0; o >>= 1) sq += __shfl_xor(sq, o, 32);
    const float inv = rsqrtf(sq * (1.0f / DD) + LN_EPS);
    const float4 gg = *(const float4*)(g + lane * 4);
    const float4 bb = *(const float4*)(b + lane * 4);
    v4bf o4;
    o4[0] = (__bf16)(c.x * inv * gg.x + bb.x);
    o4[1] = (__bf16)(c.y * inv * gg.y + bb.y);
    o4[2] = (__bf16)(c.z * inv * gg.z + bb.z);
    o4[3] = (__bf16)(c.w * inv * gg.w + bb.w);
    *(v4bf*)(normed + (size_t)node * DD + lane * 4) = o4;
}

// ---------------------------------------------------------------------------
extern "C" void kernel_launch(void* const* d_in, const int* in_sizes, int n_in,
                              void* d_out, int out_size, void* d_ws, size_t ws_size,
                              hipStream_t stream) {
    const float*     x    = (const float*)d_in[0];
    const long long* ei   = (const long long*)d_in[1];
    const long long* vp   = (const long long*)d_in[2];
    const float*     pe   = (const float*)d_in[3];
    const float*     Wq   = (const float*)d_in[4];
    const float*     Wk   = (const float*)d_in[5];
    const float*     Wout = (const float*)d_in[6];
    const float*     lng  = (const float*)d_in[7];
    const float*     lnb  = (const float*)d_in[8];
    float*           out  = (float*)d_out;

    const int N = in_sizes[0] / DD;   // 50000
    const int E = in_sizes[1] / 2;    // 800000

    // ---- carve workspace (~120 MB) ----
    char* ws = (char*)d_ws;
    auto carve = [&](size_t bytes) -> char* {
        char* p = ws;
        ws += (bytes + 255) & ~(size_t)255;
        return p;
    };
    __bf16*   xq_bf  = (__bf16*)carve((size_t)N * DD * 2);  // later: normed
    __bf16*   x_bf   = (__bf16*)carve((size_t)N * DD * 2);
    __bf16*   Wqt    = (__bf16*)carve(WBYTES);
    __bf16*   Wkt    = (__bf16*)carve(WBYTES);
    __bf16*   Wot    = (__bf16*)carve(WBYTES);
    float*    hq     = (float*)carve((size_t)N * DD * 4);   // later: agg
    float*    hk     = (float*)carve((size_t)N * DD * 4);
    float*    hv     = (float*)carve((size_t)N * DD * 4);
    float*    scores = (float*)carve((size_t)E * HH * 4);
    unsigned* segmax = (unsigned*)carve((size_t)N * HH * 4);
    float*    denom  = (float*)carve((size_t)N * HH * 4);
    int*      deg    = (int*)carve((size_t)N * 4);

    float*  agg    = hq;               // alias: hq dead after edge_scores
    __bf16* normed = xq_bf;            // alias: xq dead after QKV GEMM

    const int B = 256;
    auto cdiv = [](size_t a, size_t b) { return (unsigned)((a + b - 1) / b); };

    // 1) prep inputs + weights
    prep_x_k<<<cdiv((size_t)N * DD, B), B, 0, stream>>>(x, vp, pe, xq_bf, x_bf, N);
    prep_w_k<<<cdiv((size_t)DD * DD, B), B, 0, stream>>>(Wq, Wk, Wout, Wqt, Wkt, Wot);

    // 2) init softmax / degree state (every call — replay-safe)
    fill_u32_k<<<cdiv((size_t)N * HH, B), B, 0, stream>>>(segmax, NEG_INF_ORD, (size_t)N * HH);
    fill_u32_k<<<cdiv((size_t)N * HH, B), B, 0, stream>>>((unsigned*)denom, 0u, (size_t)N * HH);
    fill_u32_k<<<cdiv((size_t)N, B), B, 0, stream>>>((unsigned*)deg, 0u, (size_t)N);

    // 3) fused QKV projection (WMMA + async-LDS weight staging)
    const unsigned mtiles = cdiv((size_t)N, 16);
    gemm_qkv_k<<<mtiles, B, 0, stream>>>(xq_bf, x_bf, Wqt, Wkt, hq, hk, hv, N);

    // 4) edge scores + segment max + degree
    edge_scores_k<<<cdiv((size_t)E * HH, B), B, 0, stream>>>(ei, hq, hk, scores, segmax, deg, E);

    // 5) zero agg (aliases hq — hq no longer needed)
    fill_u32_k<<<cdiv((size_t)N * DD, B), B, 0, stream>>>((unsigned*)agg, 0u, (size_t)N * DD);

    // 6) exp + segment sum
    edge_exp_k<<<cdiv((size_t)E * HH, B), B, 0, stream>>>(ei, scores, segmax, denom, E);

    // 7) weighted scatter aggregation
    edge_agg_k<<<cdiv((size_t)E, B), B, 0, stream>>>(ei, scores, denom, hv, agg, E);

    // 8) LayerNorm -> bf16 (normed aliases xq_bf)
    layernorm_k<<<cdiv((size_t)N, 8), B, 0, stream>>>(agg, lng, lnb, normed, N);

    // 9) final WMMA GEMM fused with residual + isolated-node select
    gemm_out_k<<<mtiles, B, 0, stream>>>(normed, Wot, x, deg, out, N);
}
